// multi_HAN_71416716198459
// MI455X (gfx1250) — compile-verified, matrix operand order checked by reading, not run on hardware
//
#include <hip/hip_runtime.h>
#include <hip/hip_bf16.h>
#include <math.h>

#define B_SIZE  4096
#define K_NEIGH 32
#define D_DIM   256
#define DSEM    128
#define N_META  4

typedef __attribute__((ext_vector_type(16))) __bf16 v16bf;
typedef __attribute__((ext_vector_type(8)))  float  v8f;

__device__ __forceinline__ float waveSum(float v) {
#pragma unroll
    for (int o = 16; o > 0; o >>= 1) v += __shfl_xor(v, o, 32);
    return v;
}
__device__ __forceinline__ float waveMax(float v) {
#pragma unroll
    for (int o = 16; o > 0; o >>= 1) v = fmaxf(v, __shfl_xor(v, o, 32));
    return v;
}
__device__ __forceinline__ float4 add4(float4 a, float4 b) {
    return make_float4(a.x + b.x, a.y + b.y, a.z + b.z, a.w + b.w);
}
__device__ __forceinline__ float dot4(float4 a, float4 b) {
    return a.x * b.x + a.y * b.y + a.z * b.z + a.w * b.w;
}

// ---------------------------------------------------------------------------
// Kernel 1: per-metapath GAT node attention. One block per batch row b.
// 256 threads = 8 waves; wave w gathers neighbor rows 4w..4w+3 (1 KB each)
// into LDS, fusing bias add + a2-dot. Softmax over K in wave 0, then the
// alpha-weighted sum + elu is column-parallel across all 256 threads.
// ---------------------------------------------------------------------------
__global__ __launch_bounds__(256)
void han_node_attn(const int* __restrict__ users,
                   const int* __restrict__ nUser, const int* __restrict__ nBus,
                   const int* __restrict__ nCity, const int* __restrict__ nCat,
                   const float* __restrict__ Wuser, const float* __restrict__ bUser,
                   const float* __restrict__ Wbus,  const float* __restrict__ bBus,
                   const float* __restrict__ Wcity, const float* __restrict__ bCity,
                   const float* __restrict__ Wcat,  const float* __restrict__ bCat,
                   const float* __restrict__ attn,
                   float* __restrict__ metas)
{
    __shared__ float nbuf[K_NEIGH * D_DIM];   // 32 KB neighbor tile
    __shared__ float hLds[D_DIM];
    __shared__ float eLds[K_NEIGH];
    __shared__ float alphaLds[K_NEIGH];
    __shared__ float wsum[8];
    __shared__ float ha1Lds;

    const int b    = blockIdx.x;
    const int t    = threadIdx.x;
    const int wave = t >> 5;
    const int lane = t & 31;

    // target embedding h = W_user[u] + b_user
    const int u = users[b];
    hLds[t] = Wuser[(size_t)u * D_DIM + t] + bUser[t];
    __syncthreads();

    const float* Wtab[N_META] = {Wuser, Wbus, Wcity, Wcat};
    const float* btab[N_META] = {bUser, bBus, bCity, bCat};
    const int*   ntab[N_META] = {nUser, nBus, nCity, nCat};

    for (int m = 0; m < N_META; ++m) {
        const float* a1 = attn + (size_t)m * (2 * D_DIM);
        const float* a2 = a1 + D_DIM;

        // ---- h . a1 (block reduction) ----
        float p = hLds[t] * a1[t];
        p = waveSum(p);
        if (lane == 0) wsum[wave] = p;
        __syncthreads();                      // also fences nbuf reads from prev m
        if (t == 0) {
            float s = 0.f;
#pragma unroll
            for (int w = 0; w < 8; ++w) s += wsum[w];
            ha1Lds = s;
        }
        __syncthreads();
        const float ha1 = ha1Lds;

        const float* Wm = Wtab[m];
        const float* bm = btab[m];
        const int*   nm = ntab[m] + (size_t)b * K_NEIGH;

        // lane owns 8 consecutive columns: two float4 slots
        const float4 bb0 = ((const float4*)bm)[lane * 2];
        const float4 bb1 = ((const float4*)bm)[lane * 2 + 1];
        const float4 a20 = ((const float4*)a2)[lane * 2];
        const float4 a21 = ((const float4*)a2)[lane * 2 + 1];

        // ---- gather K neighbor rows, fuse bias add + a2-dot ----
#pragma unroll
        for (int j = 0; j < 4; ++j) {
            const int k  = wave * 4 + j;
            const int nb = nm[k];
            if (j < 3) {                       // prefetch the next gathered row
                const int nbn = nm[k + 1];
                __builtin_prefetch(Wm + (size_t)nbn * D_DIM, 0, 3);
            }
            const float4* src = (const float4*)(Wm + (size_t)nb * D_DIM);
            float4 x0 = add4(src[lane * 2],     bb0);
            float4 x1 = add4(src[lane * 2 + 1], bb1);
            ((float4*)(nbuf + k * D_DIM))[lane * 2]     = x0;
            ((float4*)(nbuf + k * D_DIM))[lane * 2 + 1] = x1;
            float pd = dot4(x0, a20) + dot4(x1, a21);
            pd = waveSum(pd);
            if (lane == 0) {
                float e = ha1 + pd;
                eLds[k] = e > 0.f ? e : 0.2f * e;   // leaky_relu(0.2)
            }
        }
        __syncthreads();

        // ---- softmax over K=32 (one wave, lane==k) ----
        if (wave == 0) {
            float e  = eLds[lane];
            float mx = waveMax(e);
            float ex = __expf(e - mx);
            float sm = waveSum(ex);
            alphaLds[lane] = ex / sm;
        }
        __syncthreads();

        // ---- alpha-weighted sum over neighbors + elu ----
        float acc = 0.f;
#pragma unroll 8
        for (int k = 0; k < K_NEIGH; ++k)
            acc += alphaLds[k] * nbuf[k * D_DIM + t];
        float r = acc > 0.f ? acc : expm1f(acc);    // elu(alpha=1)
        metas[((size_t)m * B_SIZE + b) * D_DIM + t] = r;
        // nbuf/eLds/alphaLds rewrite next iteration is fenced by the
        // __syncthreads() calls in the ha1 phase above.
    }
}

// ---------------------------------------------------------------------------
// Kernel 2: semantic attention GEMM [16384x256] x [256x128] on the WMMA pipe
// (bf16 in, f32 accumulate), fused with tanh(.+b_sem).q_sem and the mean-
// over-batch reduction into s[4]. One wave per 16-row M tile; A fragments
// hoisted across N tiles. Fragment layouts per CDNA5 ISA 7.12.2.
// ---------------------------------------------------------------------------
__global__ __launch_bounds__(256)
void han_sem_gemm(const float* __restrict__ metas, const float* __restrict__ Wsem,
                  const float* __restrict__ bSem,  const float* __restrict__ qSem,
                  float* __restrict__ sAcc)
{
    const int tid    = threadIdx.x;
    const int wave   = tid >> 5;
    const int lane   = tid & 31;
    const int mLoc   = lane & 15;   // M (for A) / N (for B) within tile
    const int half   = lane >> 4;
    const int rowBase = blockIdx.x * 128 + wave * 16;      // global meta row
    const int m      = rowBase >> 12;                       // 4096 rows / meta

    // A fragments for all 8 K-steps (16-bit A 16x32 layout: lane m = lane&15,
    // VGPR v<4 -> K = half*8 + 2v, v>=4 -> K = 16 + half*8 + 2(v-4))
    const float* arow = metas + (size_t)(rowBase + mLoc) * D_DIM;
    v16bf afrag[8];
#pragma unroll
    for (int ks = 0; ks < 8; ++ks) {
        const float* ap = arow + ks * 32;
#pragma unroll
        for (int v = 0; v < 8; ++v) {
            const int k = (v < 4) ? (half * 8 + 2 * v)
                                  : (16 + half * 8 + 2 * (v - 4));
            afrag[ks][2 * v]     = (__bf16)ap[k];
            afrag[ks][2 * v + 1] = (__bf16)ap[k + 1];
        }
    }

    float sPart = 0.f;
#pragma unroll
    for (int nt = 0; nt < 8; ++nt) {
        const int n = nt * 16 + mLoc;
        v8f c = {};
#pragma unroll
        for (int ks = 0; ks < 8; ++ks) {
            const float* bp = Wsem + (size_t)(ks * 32) * DSEM + n;
            v16bf bfrag;
#pragma unroll
            for (int v = 0; v < 8; ++v) {
                const int k = (v < 4) ? (half * 8 + 2 * v)
                                      : (16 + half * 8 + 2 * (v - 4));
                bfrag[2 * v]     = (__bf16)bp[(size_t)k * DSEM];
                bfrag[2 * v + 1] = (__bf16)bp[(size_t)(k + 1) * DSEM];
            }
            c = __builtin_amdgcn_wmma_f32_16x16x32_bf16(
                    false, afrag[ks], false, bfrag, (short)0, c, false, false);
        }
        // fused epilogue: tanh(c + b_sem[n]) * q_sem[n], summed over all rows
        const float bs = bSem[n], qs = qSem[n];
#pragma unroll
        for (int r = 0; r < 8; ++r) sPart += tanhf(c[r] + bs) * qs;
    }
    sPart = waveSum(sPart);
    if (lane == 0) atomicAdd(&sAcc[m], sPart);   // global_atomic_add_f32
}

// ---------------------------------------------------------------------------
// Kernel 3: beta = softmax(s / B) (4 scalars, computed redundantly) and the
// weighted combine out = sum_m beta_m * metas[m].
// ---------------------------------------------------------------------------
__global__ __launch_bounds__(256)
void han_combine(const float* __restrict__ metas, const float* __restrict__ sAcc,
                 float* __restrict__ out)
{
    const int i = blockIdx.x * 256 + threadIdx.x;          // over B*D
    const float inv = 1.0f / (float)B_SIZE;
    float s0 = sAcc[0] * inv, s1 = sAcc[1] * inv, s2 = sAcc[2] * inv, s3 = sAcc[3] * inv;
    float mx = fmaxf(fmaxf(s0, s1), fmaxf(s2, s3));
    float e0 = __expf(s0 - mx), e1 = __expf(s1 - mx);
    float e2 = __expf(s2 - mx), e3 = __expf(s3 - mx);
    const float isum = 1.0f / (e0 + e1 + e2 + e3);

    const size_t stride = (size_t)B_SIZE * D_DIM;
    float v = e0 * isum * metas[i]
            + e1 * isum * metas[i + stride]
            + e2 * isum * metas[i + 2 * stride]
            + e3 * isum * metas[i + 3 * stride];
    out[i] = v;
}

// ---------------------------------------------------------------------------
extern "C" void kernel_launch(void* const* d_in, const int* in_sizes, int n_in,
                              void* d_out, int out_size, void* d_ws, size_t ws_size,
                              hipStream_t stream)
{
    const int*   users = (const int*)d_in[0];
    /* d_in[1] businesses: unused by the reference output */
    const int*   nUser = (const int*)d_in[2];
    const int*   nBus  = (const int*)d_in[3];
    const int*   nCity = (const int*)d_in[4];
    const int*   nCat  = (const int*)d_in[5];
    const float* Wuser = (const float*)d_in[6];
    const float* bUser = (const float*)d_in[7];
    const float* Wbus  = (const float*)d_in[8];
    const float* bBus  = (const float*)d_in[9];
    const float* Wcity = (const float*)d_in[10];
    const float* bCity = (const float*)d_in[11];
    const float* Wcat  = (const float*)d_in[12];
    const float* bCat  = (const float*)d_in[13];
    const float* attn  = (const float*)d_in[14];
    const float* Wsem  = (const float*)d_in[15];
    const float* bSem  = (const float*)d_in[16];
    const float* qSem  = (const float*)d_in[17];

    float* metas = (float*)d_ws;                                   // 16 MB
    float* sAcc  = (float*)((char*)d_ws +
                            (size_t)N_META * B_SIZE * D_DIM * sizeof(float));

    hipMemsetAsync(sAcc, 0, N_META * sizeof(float), stream);

    han_node_attn<<<B_SIZE, 256, 0, stream>>>(
        users, nUser, nBus, nCity, nCat,
        Wuser, bUser, Wbus, bBus, Wcity, bCity, Wcat, bCat,
        attn, metas);

    han_sem_gemm<<<(N_META * B_SIZE) / 128, 256, 0, stream>>>(
        metas, Wsem, bSem, qSem, sAcc);

    han_combine<<<(B_SIZE * D_DIM) / 256, 256, 0, stream>>>(
        metas, sAcc, (float*)d_out);

    (void)in_sizes; (void)n_in; (void)out_size; (void)ws_size;
}
